// REGCNBase_71004399337808
// MI455X (gfx1250) — matrix operand order repaired and checked by reading, not compiled, not run on hardware
//
#include <hip/hip_runtime.h>
#include <math.h>

// ---------------- problem constants (match reference) ----------------
#define N_ENT   50000
#define TWO_R   400
#define DIM     256
#define T_STEPS 8
#define N_EDGE  100000

typedef __bf16 bf16_t;
typedef __attribute__((ext_vector_type(16))) __bf16 v16bf;
typedef __attribute__((ext_vector_type(8)))  float  v8f;
typedef unsigned int u32x4 __attribute__((ext_vector_type(4)));
typedef int          i32x8 __attribute__((ext_vector_type(8)));
typedef int          i32x4 __attribute__((ext_vector_type(4)));

// ---------------------------------------------------------------------
// Small utility kernels
// ---------------------------------------------------------------------
__global__ void k_zero_f32(float* p, int n) {
    int i = blockIdx.x * blockDim.x + threadIdx.x;
    if (i < n) p[i] = 0.f;
}

__global__ void k_copy_f32(float* dst, const float* src, int n) {
    int i = blockIdx.x * blockDim.x + threadIdx.x;
    if (i < n) dst[i] = src[i];
}

__global__ void k_cvt_bf16(bf16_t* dst, const float* src, int n) {
    int i = blockIdx.x * blockDim.x + threadIdx.x;
    if (i < n) dst[i] = (bf16_t)src[i];
}

// f32 W[K,N] row-major -> bf16 out[N,K] (B-frags contiguous in K)
__global__ void k_cvt_bf16_t(bf16_t* dst, const float* src, int K, int N) {
    int i = blockIdx.x * blockDim.x + threadIdx.x;
    if (i < K * N) {
        int k = i / N, n = i % N;
        dst[(size_t)n * K + k] = (bf16_t)src[i];
    }
}

// Row L2-normalize: one wave32 per row, 8 elems/lane
__global__ void k_l2norm(float* out, const float* in) {
    int wave = threadIdx.x >> 5, lane = threadIdx.x & 31;
    int row  = blockIdx.x * 8 + wave;
    const float* r = in + (size_t)row * DIM;
    float v[8]; float ss = 0.f;
#pragma unroll
    for (int i = 0; i < 8; ++i) { v[i] = r[lane + 32 * i]; ss += v[i] * v[i]; }
#pragma unroll
    for (int m = 16; m >= 1; m >>= 1) ss += __shfl_xor(ss, m, 32);
    float inv = 1.f / fmaxf(sqrtf(ss), 1e-12f);
    float* o = out + (size_t)row * DIM;
#pragma unroll
    for (int i = 0; i < 8; ++i) o[lane + 32 * i] = v[i] * inv;
}

// ---------------------------------------------------------------------
// Unique-(entity,relation) flag bitmap (1 byte per pair, r-major)
// ---------------------------------------------------------------------
__global__ void k_flag_mark(unsigned* flags, const int* src, const int* dst, const int* rel) {
    int i = blockIdx.x * blockDim.x + threadIdx.x;
    if (i >= 2 * N_EDGE) return;
    int e = (i < N_EDGE) ? src[i] : dst[i - N_EDGE];
    int r = (i < N_EDGE) ? rel[i] : rel[i - N_EDGE];
    unsigned idx = (unsigned)r * (unsigned)N_ENT + (unsigned)e;
    atomicOr(&flags[idx >> 2], 1u << ((idx & 3u) * 8u));
}

__global__ void k_avg_rel(float* dyn, const unsigned char* flags, const float* h) {
    int r = blockIdx.x, tid = threadIdx.x;
    __shared__ unsigned char sf[256];
    const unsigned char* fr = flags + (size_t)r * N_ENT;
    float acc = 0.f; int cnt = 0;
    for (int base = 0; base < N_ENT; base += 256) {
        int e = base + tid;
        sf[tid] = (e < N_ENT) ? fr[e] : (unsigned char)0;
        __syncthreads();
        int lim = (N_ENT - base < 256) ? (N_ENT - base) : 256;
        for (int j = 0; j < lim; ++j)
            if (sf[j]) { acc += h[(size_t)(base + j) * DIM + tid]; cnt++; }
        __syncthreads();
    }
    dyn[r * DIM + tid] = cnt ? acc / (float)cnt : 0.f;
}

__global__ void k_build_x(float* x, const float* srel, const float* dyn) {
    int i = blockIdx.x * blockDim.x + threadIdx.x;
    if (i >= TWO_R * 2 * DIM) return;
    int r = i / (2 * DIM), c = i % (2 * DIM);
    x[i] = (c < DIM) ? srel[r * DIM + c] : dyn[r * DIM + (c - DIM)];
}

__global__ void k_gru(float* relh, const float* xg, const float* hg,
                      const float* bih, const float* bhh) {
    int i = blockIdx.x * blockDim.x + threadIdx.x;
    if (i >= TWO_R * DIM) return;
    int r = i / DIM, d = i % DIM;
    const float* xr = xg + (size_t)r * 3 * DIM;
    const float* hr = hg + (size_t)r * 3 * DIM;
    float xr_ = xr[d]           + bih[d];
    float xz_ = xr[DIM + d]     + bih[DIM + d];
    float xn_ = xr[2 * DIM + d] + bih[2 * DIM + d];
    float hr_ = hr[d]           + bhh[d];
    float hz_ = hr[DIM + d]     + bhh[DIM + d];
    float hn_ = hr[2 * DIM + d] + bhh[2 * DIM + d];
    float rr = 1.f / (1.f + __expf(-(xr_ + hr_)));
    float z  = 1.f / (1.f + __expf(-(xz_ + hz_)));
    float n  = tanhf(xn_ + rr * hn_);
    float hp = relh[i];
    relh[i] = (1.f - z) * n + z * hp;
}

__global__ void k_deg(float* deg, const int* dst) {
    int i = blockIdx.x * blockDim.x + threadIdx.x;
    if (i < N_EDGE) atomicAdd(&deg[dst[i]], 1.f);
}

__global__ void k_combine(float* cur, const float* agg, const float* deg) {
    int i = blockIdx.x * blockDim.x + threadIdx.x;
    if (i >= N_ENT * DIM) return;
    int row = i / DIM;
    cur[i] = agg[i] / fmaxf(deg[row], 1.f) + cur[i];
}

__global__ void k_gate(float* out, const float* pre, const float* gbias,
                       const float* cur, const float* hprev) {
    int i = blockIdx.x * blockDim.x + threadIdx.x;
    if (i >= N_ENT * DIM) return;
    int d = i % DIM;
    float g = 1.f / (1.f + __expf(-(pre[i] + gbias[d])));
    out[i] = g * cur[i] + (1.f - g) * hprev[i];
}

// ---------------------------------------------------------------------
// TDM stage: DMA the B panel (256 columns x K bf16, stored [N,K]) into
// LDS. 2D tile, data_size=4B, row = K/2 dwords, 256 rows; pad_enable
// adds 16B after every row (pad_interval = row dwords, pad_amount = 4
// dwords) so LDS column stride = 2K+16 bytes (bank-conflict relief).
// D# layout per CDNA5 ISA.
// Toolchain uses the 6-arg builtin form:
//   (u32x4 g0, i32x8 g1, i32x4 g2, i32x4 g3, i32x8 extra, i32 cpol)
// ---------------------------------------------------------------------
template<int KT>
__device__ inline void stage_B_tdm(const bf16_t* Bt, int col0) {
    if (threadIdx.x < 32) {
        unsigned long long ga = (unsigned long long)(const void*)(Bt + (size_t)col0 * KT);
        const unsigned row_dw = KT / 2;                 // dwords per column-row
        const unsigned pi     = (KT == 256) ? 6u : 7u;  // 128 / 256 dwords
        const unsigned ncols  = 256u;
        u32x4 g0;
        g0[0] = 1u;                                     // count=1
        g0[1] = 0u;                                     // lds_addr = 0 (dyn LDS base)
        g0[2] = (unsigned)(ga & 0xffffffffu);
        g0[3] = (unsigned)((ga >> 32) & 0x01ffffffu) | (2u << 30);  // type=2
        i32x8 g1;
        g1[0] = (int)((2u << 16) | (1u << 20) | (pi << 22) | (3u << 25));
        g1[1] = (int)((row_dw & 0xffffu) << 16);                       // tdim0[15:0]
        g1[2] = (int)(((row_dw >> 16) & 0xffffu) | ((ncols & 0xffffu) << 16)); // tdim0 hi, tdim1 lo
        g1[3] = (int)(((ncols >> 16) & 0xffffu) | ((row_dw & 0xffffu) << 16)); // tdim1 hi, tile_dim0
        g1[4] = (int)(ncols & 0xffffu);                                // tile_dim1, tile_dim2=0
        g1[5] = (int)row_dw;                                           // tdim0_stride lo32
        g1[6] = 0;
        g1[7] = 0;
        i32x4 z4; z4[0] = 0; z4[1] = 0; z4[2] = 0; z4[3] = 0;
        i32x8 z8;
#pragma unroll
        for (int j = 0; j < 8; ++j) z8[j] = 0;
        __builtin_amdgcn_tensor_load_to_lds(g0, g1, z4, z4, z8, 0);
        __builtin_amdgcn_s_wait_tensorcnt(0);
    }
    __syncthreads();
}

// ---------------------------------------------------------------------
// WMMA GEMM:  C[M,N] (+)= op(A)[M,K] @ B[K,N]
//   A f32 (optional gather row index gA; FUSED adds gathered A2 rows),
//   B bf16 stored [N,K], TDM-staged per block into LDS,
//   C plain store or fused segment-sum scatter (atomicAdd via sC).
// Wave tile 16x128 (8x v8f acc), block 8 waves = 64(M) x 256(N).
// K-loop fully unrolled (templated K), A frag software-pipelined so
// global loads of step k+1 overlap the 8 WMMAs of step k.
// Requires N % 256 == 0 (256, 768 here); M guarded.
// ---------------------------------------------------------------------
template<int KT, bool FUSED>
__device__ inline v16bf load_a_frag(const float* a0, const float* a1, int kk) {
    v16bf af;
#pragma unroll
    for (int j = 0; j < 8; ++j) {
        float lo = a0[kk + j];
        float hi = a0[kk + 16 + j];
        if (FUSED) { lo += a1[kk + j]; hi += a1[kk + 16 + j]; }
        af[j]     = (bf16_t)lo;
        af[j + 8] = (bf16_t)hi;
    }
    return af;
}

template<int KT, bool FUSED>
__global__ __launch_bounds__(256)
void k_gemm(const float* __restrict__ A, const int* __restrict__ gA,
            const float* __restrict__ A2, const int* __restrict__ gA2,
            const bf16_t* __restrict__ Bt, float* __restrict__ C,
            const int* __restrict__ sC, int M, int N) {
    extern __shared__ char smem[];              // 256 * (2*KT + 16) bytes
    const int ROWB = 2 * KT + 16;               // LDS bytes per B column

    int wave = threadIdx.x >> 5;
    int lane = threadIdx.x & 31;
    int lm = lane & 15;
    int kh = (lane >> 4) << 3;                  // A frag K offset: 0 / 8
    int kb = (lane >> 4) << 3;                  // B frag K offset: 0 / 8
    int wm = blockIdx.x * 64 + (wave >> 1) * 16;
    int wnl = (wave & 1) * 128;                 // wave's local col base
    int colBase = blockIdx.y * 256;

    stage_B_tdm<KT>(Bt, colBase);

    int row  = wm + lm;
    int rowc = row < M ? row : (M - 1);
    const float* a0 = A + (size_t)(gA ? gA[rowc] : rowc) * KT + kh;
    const float* a1 = FUSED ? (A2 + (size_t)gA2[rowc] * KT + kh) : (const float*)0;

    v8f acc[8];
#pragma unroll
    for (int t = 0; t < 8; ++t)
#pragma unroll
        for (int e = 0; e < 8; ++e) acc[t][e] = 0.f;

    v16bf af = load_a_frag<KT, FUSED>(a0, a1, 0);
#pragma unroll
    for (int ks = 0; ks < KT / 32; ++ks) {
        v16bf afn;
        if (ks + 1 < KT / 32)
            afn = load_a_frag<KT, FUSED>(a0, a1, (ks + 1) * 32);
#pragma unroll
        for (int t = 0; t < 8; ++t) {
            const bf16_t* bp = (const bf16_t*)(smem +
                (size_t)(wnl + t * 16 + lm) * ROWB + (size_t)(ks * 32 + kb) * 2);
            v16bf bf;
#pragma unroll
            for (int e = 0; e < 16; ++e) bf[e] = bp[e < 8 ? e : e + 8];
            acc[t] = __builtin_amdgcn_wmma_f32_16x16x32_bf16(
                false, af, false, bf, (short)0, acc[t], false, false);
        }
        af = afn;
    }

    int m0 = wm + ((lane >> 4) << 3);           // C layout: acc[t][e] -> M = m0+e
#pragma unroll
    for (int t = 0; t < 8; ++t) {
        int col = colBase + wnl + t * 16 + lm;
        if (sC) {
#pragma unroll
            for (int e = 0; e < 8; ++e) {
                int m = m0 + e;
                if (m < M) atomicAdd(&C[(size_t)sC[m] * N + col], acc[t][e]);
            }
        } else {
#pragma unroll
            for (int e = 0; e < 8; ++e) {
                int m = m0 + e;
                if (m < M) C[(size_t)m * N + col] = acc[t][e];
            }
        }
    }
}

// ---------------------------------------------------------------------
// Host orchestration
// ---------------------------------------------------------------------
static inline int cdiv(int a, int b) { return (a + b - 1) / b; }

extern "C" void kernel_launch(void* const* d_in, const int* in_sizes, int n_in,
                              void* d_out, int out_size, void* d_ws, size_t ws_size,
                              hipStream_t stream) {
    (void)in_sizes; (void)n_in; (void)out_size; (void)ws_size;
    const int*   edge_src = (const int*)d_in[0];
    const int*   edge_rel = (const int*)d_in[1];
    const int*   edge_dst = (const int*)d_in[2];
    const float* ent      = (const float*)d_in[3];
    const float* srel     = (const float*)d_in[4];
    const float* gateW    = (const float*)d_in[5];
    const float* gateB    = (const float*)d_in[6];
    const float* w_ih     = (const float*)d_in[7];   // [768,512]
    const float* w_hh     = (const float*)d_in[8];   // [768,256]
    const float* b_ih     = (const float*)d_in[9];
    const float* b_hh     = (const float*)d_in[10];
    const float* w_neigh  = (const float*)d_in[11];  // [2,256,256]
    const float* w_loop   = (const float*)d_in[12];  // [2,256,256]

    // ---- workspace carve ----
    char* ws = (char*)d_ws; size_t off = 0;
    auto carve = [&](size_t bytes) -> void* {
        void* p = ws + off; off += (bytes + 255) & ~(size_t)255; return p;
    };
    float*    h0    = (float*)carve((size_t)N_ENT * DIM * 4);
    float*    agg   = (float*)carve((size_t)N_ENT * DIM * 4);   // reused as gate-pre
    float*    tmpA  = (float*)carve((size_t)N_ENT * DIM * 4);
    float*    tmpB  = (float*)carve((size_t)N_ENT * DIM * 4);
    float*    deg   = (float*)carve((size_t)N_ENT * 4);
    unsigned* flags = (unsigned*)carve((size_t)TWO_R * N_ENT);  // 1 byte/pair
    float*    dyn   = (float*)carve((size_t)TWO_R * DIM * 4);
    float*    x     = (float*)carve((size_t)TWO_R * 2 * DIM * 4);
    float*    xg    = (float*)carve((size_t)TWO_R * 3 * DIM * 4);
    float*    hg    = (float*)carve((size_t)TWO_R * 3 * DIM * 4);
    float*    relh  = (float*)carve((size_t)TWO_R * DIM * 4);
    bf16_t*   wbn0  = (bf16_t*)carve((size_t)DIM * DIM * 2);
    bf16_t*   wbn1  = (bf16_t*)carve((size_t)DIM * DIM * 2);
    bf16_t*   wbl0  = (bf16_t*)carve((size_t)DIM * DIM * 2);
    bf16_t*   wbl1  = (bf16_t*)carve((size_t)DIM * DIM * 2);
    bf16_t*   wbg   = (bf16_t*)carve((size_t)DIM * DIM * 2);
    bf16_t*   wbih  = (bf16_t*)carve((size_t)3 * DIM * 2 * DIM * 2);
    bf16_t*   wbhh  = (bf16_t*)carve((size_t)3 * DIM * DIM * 2);

    // LDS bytes for a staged 256-col B panel
    const int smem256 = 256 * (2 * 256 + 16);   // K=256
    const int smem512 = 256 * (2 * 512 + 16);   // K=512

    auto gemm256 = [&](const float* A, const bf16_t* Bt, float* C, int M, int N) {
        k_gemm<256, false><<<dim3(cdiv(M, 64), N / 256), dim3(256), smem256, stream>>>(
            A, 0, 0, 0, Bt, C, 0, M, N);
    };
    auto gemm256_fused = [&](const float* A, const int* gA, const float* A2, const int* gA2,
                             const bf16_t* Bt, float* C, const int* sC, int M, int N) {
        k_gemm<256, true><<<dim3(cdiv(M, 64), N / 256), dim3(256), smem256, stream>>>(
            A, gA, A2, gA2, Bt, C, sC, M, N);
    };
    auto gemm512 = [&](const float* A, const bf16_t* Bt, float* C, int M, int N) {
        k_gemm<512, false><<<dim3(cdiv(M, 64), N / 256), dim3(256), smem512, stream>>>(
            A, 0, 0, 0, Bt, C, 0, M, N);
    };

    // ---- one-time prep (re-run every call; deterministic) ----
    k_cvt_bf16_t<<<cdiv(DIM * DIM, 256), 256, 0, stream>>>(wbn0, w_neigh,             DIM, DIM);
    k_cvt_bf16_t<<<cdiv(DIM * DIM, 256), 256, 0, stream>>>(wbn1, w_neigh + DIM * DIM, DIM, DIM);
    k_cvt_bf16_t<<<cdiv(DIM * DIM, 256), 256, 0, stream>>>(wbl0, w_loop,              DIM, DIM);
    k_cvt_bf16_t<<<cdiv(DIM * DIM, 256), 256, 0, stream>>>(wbl1, w_loop + DIM * DIM,  DIM, DIM);
    k_cvt_bf16_t<<<cdiv(DIM * DIM, 256), 256, 0, stream>>>(wbg,  gateW,               DIM, DIM);
    k_cvt_bf16<<<cdiv(3 * DIM * 2 * DIM, 256), 256, 0, stream>>>(wbih, w_ih, 3 * DIM * 2 * DIM);
    k_cvt_bf16<<<cdiv(3 * DIM * DIM, 256), 256, 0, stream>>>(wbhh, w_hh, 3 * DIM * DIM);
    k_copy_f32<<<cdiv(TWO_R * DIM, 256), 256, 0, stream>>>(relh, srel, TWO_R * DIM);
    k_l2norm<<<N_ENT / 8, 256, 0, stream>>>(h0, ent);

    const int ND = N_ENT * DIM;
    for (int t = 0; t < T_STEPS; ++t) {
        const float* hprev = (t == 0) ? h0 : (float*)d_out + (size_t)(t - 1) * ND;
        const int* src = edge_src + (size_t)t * N_EDGE;
        const int* rel = edge_rel + (size_t)t * N_EDGE;
        const int* dst = edge_dst + (size_t)t * N_EDGE;

        // ---- dynamic relation embedding (unique (e,r) mean) ----
        k_zero_f32<<<cdiv(TWO_R * N_ENT / 4, 256), 256, 0, stream>>>((float*)flags, TWO_R * N_ENT / 4);
        k_flag_mark<<<cdiv(2 * N_EDGE, 256), 256, 0, stream>>>(flags, src, dst, rel);
        k_avg_rel<<<TWO_R, 256, 0, stream>>>(dyn, (const unsigned char*)flags, hprev);

        // ---- GRU relation update ----
        k_build_x<<<cdiv(TWO_R * 2 * DIM, 256), 256, 0, stream>>>(x, srel, dyn);
        gemm512(x,    wbih, xg, TWO_R, 3 * DIM);
        gemm256(relh, wbhh, hg, TWO_R, 3 * DIM);
        k_gru<<<cdiv(TWO_R * DIM, 256), 256, 0, stream>>>(relh, xg, hg, b_ih, b_hh);

        // ---- degree (shared by both layers) ----
        k_zero_f32<<<cdiv(N_ENT, 256), 256, 0, stream>>>(deg, N_ENT);
        k_deg<<<cdiv(N_EDGE, 256), 256, 0, stream>>>(deg, dst);

        // ---- RGCN layer 0: fused gather + GEMM + scatter-add ----
        k_zero_f32<<<cdiv(ND, 256), 256, 0, stream>>>(agg, ND);
        gemm256_fused(hprev, src, relh, rel, wbn0, agg, dst, N_EDGE, DIM);
        gemm256(hprev, wbl0, tmpA, N_ENT, DIM);
        k_combine<<<cdiv(ND, 256), 256, 0, stream>>>(tmpA, agg, deg);

        // ---- RGCN layer 1 ----
        k_zero_f32<<<cdiv(ND, 256), 256, 0, stream>>>(agg, ND);
        gemm256_fused(tmpA, src, relh, rel, wbn1, agg, dst, N_EDGE, DIM);
        gemm256(tmpA, wbl1, tmpB, N_ENT, DIM);
        k_combine<<<cdiv(ND, 256), 256, 0, stream>>>(tmpB, agg, deg);

        // ---- gate + blend (agg reused as gate pre-activation) ----
        gemm256(hprev, wbg, agg, N_ENT, DIM);
        float* out_t = (float*)d_out + (size_t)t * ND;
        k_gate<<<cdiv(ND, 256), 256, 0, stream>>>(out_t, agg, gateB, tmpB, hprev);
    }

    // final rel_h -> tail of d_out
    k_copy_f32<<<cdiv(TWO_R * DIM, 256), 256, 0, stream>>>(
        (float*)d_out + (size_t)T_STEPS * ND, relh, TWO_R * DIM);
}
